// MoEWithAdaptiveGate_35613868818876
// MI455X (gfx1250) — matrix operand order repaired and checked by reading, not compiled, not run on hardware
//
#include <hip/hip_runtime.h>

#define B_  8192
#define D_  2048
#define E_  8
#define H_  128
#define K2_ 1056   // E*H + 32 (gates + b3 folded into the final GEMM's K)

typedef __attribute__((ext_vector_type(16))) __bf16 v16bf;
typedef __attribute__((ext_vector_type(8)))  __bf16 v8bf;
typedef __attribute__((ext_vector_type(8)))  float  v8f;
typedef __attribute__((ext_vector_type(4)))  int    v4i;

// ---- CDNA5 async global->LDS path (ASYNCcnt), with safe fallback ----------
#if defined(__gfx1250__) && __has_builtin(__builtin_amdgcn_global_load_async_to_lds_b128)
#define USE_ASYNC 1
#else
#define USE_ASYNC 0
#endif

#if USE_ASYNC
typedef __attribute__((address_space(1))) v4i* gv4i_p;   // global int4*
typedef __attribute__((address_space(3))) v4i* lv4i_p;   // LDS int4*
#endif

static __device__ __forceinline__ void cp_async16(void* lds, const void* g) {
#if USE_ASYNC
    __builtin_amdgcn_global_load_async_to_lds_b128(
        (gv4i_p)(v4i*)(void*)g, (lv4i_p)(v4i*)lds, 0, 0);
#else
    *(uint4*)lds = *(const uint4*)g;
#endif
}
static __device__ __forceinline__ void wait_async() {
#if USE_ASYNC
#if __has_builtin(__builtin_amdgcn_s_wait_asynccnt)
    __builtin_amdgcn_s_wait_asynccnt(0);
#else
    asm volatile("s_wait_asynccnt 0x0" ::: "memory");
#endif
#endif
}

static __device__ __forceinline__ __bf16 f2bf(float f) {
    unsigned u = __builtin_bit_cast(unsigned, f);
    u += 0x7FFFu + ((u >> 16) & 1u);            // round-to-nearest-even
    unsigned short s = (unsigned short)(u >> 16);
    return __builtin_bit_cast(__bf16, s);
}
static __device__ __forceinline__ unsigned pack2(float a, float b) {
    unsigned lo = __builtin_bit_cast(unsigned short, f2bf(a));
    unsigned hi = __builtin_bit_cast(unsigned short, f2bf(b));
    return lo | (hi << 16);
}
static __device__ __forceinline__ float silu(float v) {
    return v / (1.0f + __expf(-v));
}

// ---------------------------------------------------------------- conversions
__global__ __launch_bounds__(256) void k_cvt_x(const float* __restrict__ x,
                                               __bf16* __restrict__ xb) {
    size_t i = ((size_t)blockIdx.x * 256 + threadIdx.x) * 4;
    float4 v = *(const float4*)(x + i);
    uint2 o; o.x = pack2(v.x, v.y); o.y = pack2(v.z, v.w);
    *(uint2*)(xb + i) = o;
}

// W1 [E,D,H] f32 -> W1T [E,H,D] bf16  (N-major so B-fragments are contiguous-K)
__global__ __launch_bounds__(256) void k_cvt_w1(const float* __restrict__ w1,
                                                __bf16* __restrict__ w1t) {
    int i = blockIdx.x * 256 + threadIdx.x;       // over E*D*H
    int h = i % H_, d = (i / H_) % D_, e = i / (D_ * H_);
    w1t[((size_t)e * H_ + h) * D_ + d] = f2bf(w1[i]);
}

// W2 [E,Hin,Hout] f32 -> W2T [E,Hout,Hin] bf16
__global__ __launch_bounds__(256) void k_cvt_w2(const float* __restrict__ w2,
                                                __bf16* __restrict__ w2t) {
    int i = blockIdx.x * 256 + threadIdx.x;       // over E*H*H
    int ho = i % H_, hin = (i / H_) % H_, e = i / (H_ * H_);
    w2t[((size_t)e * H_ + ho) * H_ + hin] = f2bf(w2[i]);
}

// W3 [E,H,D] f32 + b3 [E,D] -> W3T [D, K2] bf16 : k<1024 -> W3[e,h,d];
// 1024..1031 -> b3[e,d] (paired with gate columns of G2); rest zero.
__global__ __launch_bounds__(256) void k_cvt_w3(const float* __restrict__ w3,
                                                const float* __restrict__ b3,
                                                __bf16* __restrict__ w3t) {
    int i = blockIdx.x * 256 + threadIdx.x;       // over D_*K2_
    if (i >= D_ * K2_) return;
    int k = i % K2_, d = i / K2_;
    float v;
    if (k < E_ * H_) {
        int e = k >> 7, h = k & (H_ - 1);
        v = w3[((size_t)e * H_ + h) * D_ + d];
    } else if (k < E_ * H_ + E_) {
        v = b3[(size_t)(k - E_ * H_) * D_ + d];
    } else {
        v = 0.0f;
    }
    w3t[i] = f2bf(v);
}

// ---------------------------------------------------------------- gates
// One wave per row: logits = x@gate_w + gate_b, softmax over E=8.
__global__ __launch_bounds__(256) void k_gates(const float* __restrict__ x,
                                               const float* __restrict__ gw,
                                               const float* __restrict__ gb,
                                               float* __restrict__ gates,
                                               __bf16* __restrict__ g2) {
    const int lane = threadIdx.x & 31, wave = threadIdx.x >> 5;
    const int row  = blockIdx.x * 8 + wave;
    float acc[E_] = {0.f,0.f,0.f,0.f,0.f,0.f,0.f,0.f};
    const float* xr = x + (size_t)row * D_;
    for (int d = lane; d < D_; d += 32) {
        float xv = xr[d];
        const float4* w = (const float4*)(gw + (size_t)d * E_);
        float4 w0 = w[0], w1 = w[1];
        acc[0] += xv * w0.x; acc[1] += xv * w0.y;
        acc[2] += xv * w0.z; acc[3] += xv * w0.w;
        acc[4] += xv * w1.x; acc[5] += xv * w1.y;
        acc[6] += xv * w1.z; acc[7] += xv * w1.w;
    }
    #pragma unroll
    for (int e = 0; e < E_; ++e)
        #pragma unroll
        for (int off = 16; off > 0; off >>= 1)
            acc[e] += __shfl_xor(acc[e], off, 32);
    float mx = -3.0e38f;
    #pragma unroll
    for (int e = 0; e < E_; ++e) { acc[e] += gb[e]; mx = fmaxf(mx, acc[e]); }
    float s = 0.f;
    #pragma unroll
    for (int e = 0; e < E_; ++e) { acc[e] = __expf(acc[e] - mx); s += acc[e]; }
    float inv = 1.0f / s;
    if (lane < E_) gates[(size_t)row * E_ + lane] = acc[lane] * inv;
    // gate columns (K = 1024..1031) and zero padding (1032..1055) of G2
    __bf16 gv = (lane < E_) ? f2bf(acc[lane] * inv) : f2bf(0.0f);
    g2[(size_t)row * K2_ + E_ * H_ + lane] = gv;
}

// ---------------------------------------------------------------- expert body
// grid (B/128, E). Computes G2[rows, e*H:(e+1)*H] =
//   gates[:,e] * silu(silu(x@W1[e]+b1)@W2[e]+b2)   (bf16)
__global__ __launch_bounds__(256) void k_expert(const __bf16* __restrict__ xb,
                                                const __bf16* __restrict__ w1t,
                                                const __bf16* __restrict__ w2t,
                                                const float*  __restrict__ b1,
                                                const float*  __restrict__ b2,
                                                const float*  __restrict__ gates,
                                                __bf16* __restrict__ g2) {
    const int e    = blockIdx.y;
    const int row0 = blockIdx.x * 128;
    const int tid  = threadIdx.x;
    const int lane = tid & 31, wave = tid >> 5;
    const int ln   = lane & 15, hi = lane >> 4;

    // h1 buffer aliases the (dead after stage 1) double-buffered staging tiles
    __shared__ __align__(16) union SM {
        struct { __bf16 a[2][128 * 40]; __bf16 b[2][128 * 40]; } s;  // 40 KB
        __bf16 h1[128 * 136];                                        // 34 KB
    } sm;

    const v8f vzero = {};
    v8f acc[8];
    #pragma unroll
    for (int i = 0; i < 8; ++i) acc[i] = vzero;

    // A fragment: lanes 0-15 -> M rows, K {0-15}; 16-31 -> K {16-31}
    const int arow  = wave * 16 + ln;
    const int khalf = hi * 16;         // 0 or 16 (elements)
    const int bk    = hi * 16;         // B frag: 16 consecutive K per lane half

    const int lr = tid >> 1;           // staging: row 0..127
    const int lc = (tid & 1) * 16;     // staging: 16-element column block

    const __bf16* w1e = w1t + (size_t)e * H_ * D_;
    const __bf16* w2e = w2t + (size_t)e * H_ * H_;

    // ---- stage 1: h1 = silu(x @ W1[e] + b1[e]) over K = D, double-buffered
    const __bf16* gx = xb  + (size_t)(row0 + lr) * D_ + lc;
    const __bf16* gw = w1e + (size_t)lr * D_ + lc;

    #define STAGE1(ks, buf)                                                   \
        do {                                                                  \
            const int kb_ = (ks) * 32;                                        \
            __bf16* da_ = sm.s.a[buf] + lr * 40 + lc;                         \
            __bf16* db_ = sm.s.b[buf] + lr * 40 + lc;                         \
            cp_async16(da_,     gx + kb_);                                    \
            cp_async16(da_ + 8, gx + kb_ + 8);                                \
            cp_async16(db_,     gw + kb_);                                    \
            cp_async16(db_ + 8, gw + kb_ + 8);                                \
        } while (0)

    STAGE1(0, 0);
    wait_async();
    __syncthreads();
    for (int ks = 0; ks < D_ / 32; ++ks) {
        const int buf = ks & 1;
        if (ks + 1 < D_ / 32) STAGE1(ks + 1, buf ^ 1);
        union { v16bf v; v8bf h[2]; } af, bf_;
        af.h[0] = *(const v8bf*)(sm.s.a[buf] + arow * 40 + khalf);
        af.h[1] = *(const v8bf*)(sm.s.a[buf] + arow * 40 + khalf + 8);
        #pragma unroll
        for (int nt = 0; nt < 8; ++nt) {
            const int bn = nt * 16 + ln;
            bf_.h[0] = *(const v8bf*)(sm.s.b[buf] + bn * 40 + bk);
            bf_.h[1] = *(const v8bf*)(sm.s.b[buf] + bn * 40 + bk + 8);
            acc[nt] = __builtin_amdgcn_wmma_f32_16x16x32_bf16(
                false, af.v, false, bf_.v, (short)0, acc[nt], false, false);
        }
        wait_async();
        __syncthreads();
    }
    #undef STAGE1

    // bias + silu -> sm.h1 (bf16, [128][136]); staging buffers are dead now
    #pragma unroll
    for (int nt = 0; nt < 8; ++nt) {
        const int n = nt * 16 + ln;
        const float bias = b1[e * H_ + n];
        #pragma unroll
        for (int r = 0; r < 8; ++r) {
            const int m = wave * 16 + r + 8 * hi;
            sm.h1[m * 136 + n] = f2bf(silu(acc[nt][r] + bias));
        }
    }
    __syncthreads();

    // ---- stage 2: h2 = silu(h1 @ W2[e] + b2[e]); B-frags direct from L2-resident W2T
    v8f acc2[8];
    #pragma unroll
    for (int i = 0; i < 8; ++i) acc2[i] = vzero;
    #pragma unroll
    for (int ks = 0; ks < H_ / 32; ++ks) {
        union { v16bf v; v8bf h[2]; } af, bf_;
        af.h[0] = *(const v8bf*)(sm.h1 + arow * 136 + ks * 32 + khalf);
        af.h[1] = *(const v8bf*)(sm.h1 + arow * 136 + ks * 32 + khalf + 8);
        #pragma unroll
        for (int nt = 0; nt < 8; ++nt) {
            const int bn = nt * 16 + ln;
            bf_.h[0] = *(const v8bf*)(w2e + bn * H_ + ks * 32 + bk);
            bf_.h[1] = *(const v8bf*)(w2e + bn * H_ + ks * 32 + bk + 8);
            acc2[nt] = __builtin_amdgcn_wmma_f32_16x16x32_bf16(
                false, af.v, false, bf_.v, (short)0, acc2[nt], false, false);
        }
    }

    // bias + silu, scale by gate, emit bf16 into G2[:, e*H ...]
    float gv[8];
    #pragma unroll
    for (int r = 0; r < 8; ++r)
        gv[r] = gates[(size_t)(row0 + wave * 16 + r + 8 * hi) * E_ + e];
    #pragma unroll
    for (int nt = 0; nt < 8; ++nt) {
        const int n = nt * 16 + ln;
        const float bias = b2[e * H_ + n];
        #pragma unroll
        for (int r = 0; r < 8; ++r) {
            const int row = row0 + wave * 16 + r + 8 * hi;
            float v = silu(acc2[nt][r] + bias) * gv[r];
            g2[(size_t)row * K2_ + e * H_ + n] = f2bf(v);
        }
    }
}

// ---------------------------------------------------------------- out = G2 @ W3T^T
// grid (B/128, D/128). K = 1056 includes the gate*b3 correction automatically.
__global__ __launch_bounds__(256) void k_out(const __bf16* __restrict__ g2,
                                             const __bf16* __restrict__ w3t,
                                             float* __restrict__ out) {
    const int row0 = blockIdx.x * 128;
    const int col0 = blockIdx.y * 128;
    const int tid  = threadIdx.x;
    const int lane = tid & 31, wave = tid >> 5;
    const int ln   = lane & 15, hi = lane >> 4;

    __shared__ __align__(16) __bf16 sA[2][128 * 40];
    __shared__ __align__(16) __bf16 sB[2][128 * 40];

    const v8f vzero = {};
    v8f acc[8];
    #pragma unroll
    for (int i = 0; i < 8; ++i) acc[i] = vzero;

    const int arow  = wave * 16 + ln;
    const int khalf = hi * 16;
    const int bk    = hi * 16;
    const int lr = tid >> 1;
    const int lc = (tid & 1) * 16;

    const __bf16* ga  = g2  + (size_t)(row0 + lr) * K2_ + lc;
    const __bf16* gbp = w3t + (size_t)(col0 + lr) * K2_ + lc;

    #define STAGEO(ks, buf)                                                   \
        do {                                                                  \
            const int kb_ = (ks) * 32;                                        \
            __bf16* da_ = sA[buf] + lr * 40 + lc;                             \
            __bf16* db_ = sB[buf] + lr * 40 + lc;                             \
            cp_async16(da_,     ga + kb_);                                    \
            cp_async16(da_ + 8, ga + kb_ + 8);                                \
            cp_async16(db_,     gbp + kb_);                                   \
            cp_async16(db_ + 8, gbp + kb_ + 8);                               \
        } while (0)

    STAGEO(0, 0);
    wait_async();
    __syncthreads();
    for (int ks = 0; ks < K2_ / 32; ++ks) {
        const int buf = ks & 1;
        if (ks + 1 < K2_ / 32) STAGEO(ks + 1, buf ^ 1);
        union { v16bf v; v8bf h[2]; } af, bf_;
        af.h[0] = *(const v8bf*)(sA[buf] + arow * 40 + khalf);
        af.h[1] = *(const v8bf*)(sA[buf] + arow * 40 + khalf + 8);
        #pragma unroll
        for (int nt = 0; nt < 8; ++nt) {
            const int bn = nt * 16 + ln;
            bf_.h[0] = *(const v8bf*)(sB[buf] + bn * 40 + bk);
            bf_.h[1] = *(const v8bf*)(sB[buf] + bn * 40 + bk + 8);
            acc[nt] = __builtin_amdgcn_wmma_f32_16x16x32_bf16(
                false, af.v, false, bf_.v, (short)0, acc[nt], false, false);
        }
        wait_async();
        __syncthreads();
    }
    #undef STAGEO

    #pragma unroll
    for (int nt = 0; nt < 8; ++nt) {
        const int n = col0 + nt * 16 + ln;
        #pragma unroll
        for (int r = 0; r < 8; ++r) {
            const int row = row0 + wave * 16 + r + 8 * hi;
            out[(size_t)row * D_ + n] = acc[nt][r];
        }
    }
}

// ---------------------------------------------------------------- launcher
extern "C" void kernel_launch(void* const* d_in, const int* in_sizes, int n_in,
                              void* d_out, int out_size, void* d_ws, size_t ws_size,
                              hipStream_t stream) {
    const float* x  = (const float*)d_in[0];
    const float* gw = (const float*)d_in[1];
    const float* gb = (const float*)d_in[2];
    const float* W1 = (const float*)d_in[3];
    const float* b1 = (const float*)d_in[4];
    const float* W2 = (const float*)d_in[5];
    const float* b2 = (const float*)d_in[6];
    const float* W3 = (const float*)d_in[7];
    const float* b3 = (const float*)d_in[8];
    float* out = (float*)d_out;

    char* ws = (char*)d_ws;
    __bf16* xb    = (__bf16*)(ws);                       // 32 MB
    __bf16* w1t   = (__bf16*)(ws + 33554432);            //  4 MB
    __bf16* w2t   = (__bf16*)(ws + 37748736);            // 256 KB
    __bf16* w3t   = (__bf16*)(ws + 38010880);            // ~4.1 MB
    float*  gates = (float*) (ws + 42336256);            // 256 KB
    __bf16* g2    = (__bf16*)(ws + 42598400);            // ~16.5 MB

    k_cvt_x <<<(B_ * (size_t)D_ / 4) / 256, 256, 0, stream>>>(x, xb);
    k_cvt_w1<<<(E_ * D_ * H_) / 256,        256, 0, stream>>>(W1, w1t);
    k_cvt_w2<<<(E_ * H_ * H_) / 256,        256, 0, stream>>>(W2, w2t);
    k_cvt_w3<<<(D_ * K2_ + 255) / 256,      256, 0, stream>>>(W3, b3, w3t);
    k_gates <<<B_ / 8,                      256, 0, stream>>>(x, gw, gb, gates, g2);
    k_expert<<<dim3(B_ / 128, E_),          256, 0, stream>>>(xb, w1t, w2t, b1, b2, gates, g2);
    k_out   <<<dim3(B_ / 128, D_ / 128),    256, 0, stream>>>(g2, w3t, out);
}
